// SequentialMultiHeadAttention_9113920602543
// MI455X (gfx1250) — matrix operand order repaired and checked by reading, not compiled, not run on hardware
//
#include <hip/hip_runtime.h>
#include <hip/hip_bf16.h>

#define D_MODEL 1024
#define SEQ     2048
#define NHEAD   16
#define DEPTH   64

typedef __attribute__((ext_vector_type(16))) __bf16 v16bf;
typedef __attribute__((ext_vector_type(8)))  __bf16 v8bf;
typedef __attribute__((ext_vector_type(8)))  float  v8f;
typedef __attribute__((ext_vector_type(4)))  float  f32x4;
typedef __attribute__((ext_vector_type(4)))  int    i32x4;

#if defined(__HIP_DEVICE_COMPILE__) && \
    __has_builtin(__builtin_amdgcn_global_load_async_to_lds_b128) && \
    __has_builtin(__builtin_amdgcn_s_wait_asynccnt)
#define HAVE_ASYNC_LDS 1
typedef __attribute__((address_space(1))) i32x4 as1_i32x4;
typedef __attribute__((address_space(3))) i32x4 as3_i32x4;
#else
#define HAVE_ASYNC_LDS 0
#endif

static __device__ __forceinline__ v8f wmma_bf16(v16bf a, v16bf b, v8f c) {
  return __builtin_amdgcn_wmma_f32_16x16x32_bf16(false, a, false, b, (short)0, c,
                                                 false, false);
}

// 16-byte global -> LDS copy: async (ASYNCcnt-tracked) when available.
static __device__ __forceinline__ void copy16_g2l(void* lds_dst, const void* gsrc) {
#if HAVE_ASYNC_LDS
  __builtin_amdgcn_global_load_async_to_lds_b128(
      (as1_i32x4*)gsrc, (as3_i32x4*)lds_dst, 0, 0);
#else
  *(v8bf*)lds_dst = *(const v8bf*)gsrc;
#endif
}

// ---------------------------------------------------------------------------
// proj_gemm: out[m,n] = sum_k A[m,k] * W[n,k] + bias[n]
// One wave computes a 16x64 tile (4 WMMAs sharing one A fragment per K-step).
// Block = 8 waves = 16 rows x 512 cols.
//   USE_PE:   A = f32 x + pe[batch] converted to bf16 on the fly; else A bf16.
//   OUT_MODE: 0 -> bf16 (B,H,S,64); 1 -> bf16 (B,H,64,S); 2 -> f32 (B*S, D)
// ---------------------------------------------------------------------------
template<bool USE_PE, int OUT_MODE>
__global__ __launch_bounds__(256) void proj_gemm(
    const void* __restrict__ Xv, const float* __restrict__ pe,
    const float* __restrict__ W, const float* __restrict__ bias,
    void* __restrict__ outv)
{
  const int lane = threadIdx.x & 31;
  const int wave = threadIdx.x >> 5;
  const int col  = lane & 15;
  const int half = lane >> 4;
  const int h8   = half * 8;

  const int m0 = blockIdx.y * 16;
  const int n0 = blockIdx.x * 512 + wave * 64;
  const int am = m0 + col;                       // A-fragment row for this lane

  const float* per = USE_PE ? (pe + (size_t)(m0 / SEQ) * D_MODEL) : nullptr;

  v8f acc[4] = {{}, {}, {}, {}};
  for (int k0 = 0; k0 < D_MODEL; k0 += 32) {
    // ---- A fragment (16x32 bf16): per lane two 8-element runs ----
    v16bf a;
    if (USE_PE) {
      const float* xr = (const float*)Xv + (size_t)am * D_MODEL;
      #pragma unroll
      for (int g = 0; g < 2; ++g) {
        const int kb = k0 + g * 16 + h8;
        f32x4 x0 = *(const f32x4*)(xr  + kb);
        f32x4 x1 = *(const f32x4*)(xr  + kb + 4);
        f32x4 p0 = *(const f32x4*)(per + kb);
        f32x4 p1 = *(const f32x4*)(per + kb + 4);
        #pragma unroll
        for (int e = 0; e < 4; ++e) {
          a[g * 8 + e]     = (__bf16)(x0[e] + p0[e]);
          a[g * 8 + 4 + e] = (__bf16)(x1[e] + p1[e]);
        }
      }
    } else {
      const __bf16* xr = (const __bf16*)Xv + (size_t)am * D_MODEL;
      #pragma unroll
      for (int g = 0; g < 2; ++g) {
        v8bf r = *(const v8bf*)(xr + k0 + g * 16 + h8);
        #pragma unroll
        for (int e = 0; e < 8; ++e) a[g * 8 + e] = r[e];
      }
    }

    // ---- 4 B fragments (32x16 bf16 each), A reused across all 4 WMMAs ----
    #pragma unroll
    for (int nt = 0; nt < 4; ++nt) {
      const int bn = n0 + nt * 16 + col;         // B column == weight row
      v16bf bfr;
      const float* wr = W + (size_t)bn * D_MODEL + k0 + h8;
      #pragma unroll
      for (int q4 = 0; q4 < 4; ++q4) {
        f32x4 w = *(const f32x4*)(wr + (q4 >> 1) * 16 + (q4 & 1) * 4);
        #pragma unroll
        for (int e = 0; e < 4; ++e) bfr[q4 * 4 + e] = (__bf16)w[e];
      }
      acc[nt] = wmma_bf16(a, bfr, acc[nt]);
    }
  }

  #pragma unroll
  for (int nt = 0; nt < 4; ++nt) {
    const int bn = n0 + nt * 16 + col;
    const float bv = bias[bn];
    const int hh = bn / DEPTH, dd = bn % DEPTH;
    #pragma unroll
    for (int r = 0; r < 8; ++r) {
      const float val = acc[nt][r] + bv;
      const int m  = m0 + r + 8 * half;          // C layout: row r+8*half
      const int bb = m / SEQ, ss = m % SEQ;
      if (OUT_MODE == 0) {
        ((__bf16*)outv)[(((size_t)bb * NHEAD + hh) * SEQ + ss) * DEPTH + dd] =
            (__bf16)val;
      } else if (OUT_MODE == 1) {
        ((__bf16*)outv)[(((size_t)bb * NHEAD + hh) * DEPTH + dd) * SEQ + ss] =
            (__bf16)val;
      } else {
        ((float*)outv)[(size_t)m * D_MODEL + bn] = val;
      }
    }
  }
}

// ---------------------------------------------------------------------------
// flash_attn: a block owns 8 consecutive 16-query tiles of ONE (batch, head);
// the 32-key x 64-depth K tile and 64x32 V^T tile are async-staged into LDS
// once per step and shared by all 8 waves (double-buffered, ASYNCcnt-paced).
// ---------------------------------------------------------------------------
__global__ __launch_bounds__(256) void flash_attn(
    const __bf16* __restrict__ Qh, const __bf16* __restrict__ Kh,
    const __bf16* __restrict__ Vt, __bf16* __restrict__ AO)
{
  __shared__ __bf16 Kt[2][32][DEPTH];     // [buf][key][depth]      8 KB
  __shared__ __bf16 Vs[2][DEPTH][32];     // [buf][dim][key]        8 KB
  __shared__ __bf16 Pl[8][16][32];        // per-wave P transpose   8 KB

  const int lane = threadIdx.x & 31;
  const int wave = threadIdx.x >> 5;
  const int col  = lane & 15;
  const int half = lane >> 4;
  const int h8   = half * 8;

  const int bt = blockIdx.x & 15;                 // 16 blocks per (b,h)
  const int hh = (blockIdx.x >> 4) & (NHEAD - 1);
  const int bb = blockIdx.x >> 8;
  const int qt = bt * 8 + wave;                   // this wave's query tile

  const __bf16* Qb = Qh + ((size_t)bb * NHEAD + hh) * SEQ * DEPTH;
  const __bf16* Kb = Kh + ((size_t)bb * NHEAD + hh) * SEQ * DEPTH;
  const __bf16* Vb = Vt + ((size_t)bb * NHEAD + hh) * DEPTH * SEQ;

  // cooperative stage of one 32-key step into LDS buffer `buf`
  auto stage = [&](int buf, int kt0) {
    const int t = threadIdx.x;
    {
      const int row = t >> 3, ch = (t & 7) * 8;   // 32 rows x 128B
      copy16_g2l(&Kt[buf][row][ch], Kb + (size_t)(kt0 + row) * DEPTH + ch);
    }
    {
      const int row = t >> 2, ch = (t & 3) * 8;   // 64 rows x 64B
      copy16_g2l(&Vs[buf][row][ch], Vb + (size_t)row * SEQ + kt0 + ch);
    }
  };

  // Q tile 16x64 as two A-fragments (depth halves 0..31, 32..63)
  v16bf qa[2];
  {
    const __bf16* qr = Qb + (size_t)(qt * 16 + col) * DEPTH;
    #pragma unroll
    for (int kk = 0; kk < 2; ++kk)
      #pragma unroll
      for (int g = 0; g < 2; ++g) {
        v8bf r = *(const v8bf*)(qr + kk * 32 + g * 16 + h8);
        #pragma unroll
        for (int e = 0; e < 8; ++e) qa[kk][g * 8 + e] = r[e];
      }
  }

  v8f o0 = {}, o1 = {}, o2 = {}, o3 = {};
  float mrow[8], lrow[8];
  #pragma unroll
  for (int r = 0; r < 8; ++r) { mrow[r] = -1e30f; lrow[r] = 0.0f; }

  stage(0, 0);                                    // prologue: tile 0

  for (int kt = 0; kt < SEQ; kt += 32) {
    const int buf = (kt >> 5) & 1;
    if (kt + 32 < SEQ) {
      stage(buf ^ 1, kt + 32);                    // prefetch next tile
#if HAVE_ASYNC_LDS
      __builtin_amdgcn_s_wait_asynccnt(2);        // oldest tile complete
#endif
    } else {
#if HAVE_ASYNC_LDS
      __builtin_amdgcn_s_wait_asynccnt(0);
#endif
    }
    __syncthreads();                              // tile visible to all waves

    // ---- logits 16x32 from LDS K tile ----
    v8f c[2] = {{}, {}};
    #pragma unroll
    for (int nt = 0; nt < 2; ++nt) {
      #pragma unroll
      for (int kk = 0; kk < 2; ++kk) {
        v16bf bf;
        const __bf16* p = &Kt[buf][nt * 16 + col][kk * 32 + h8];
        v8bf r0 = *(const v8bf*)(p);
        v8bf r1 = *(const v8bf*)(p + 16);
        #pragma unroll
        for (int e = 0; e < 8; ++e) { bf[e] = r0[e]; bf[8 + e] = r1[e]; }
        c[nt] = wmma_bf16(qa[kk], bf, c[nt]);
      }
    }

    // ---- online softmax (rows striped across 16-lane halves) ----
    const float scale = 0.125f;                   // 1/sqrt(64)
    float tmax[8], alpha[8], psum[8];
    #pragma unroll
    for (int r = 0; r < 8; ++r) {
      c[0][r] *= scale; c[1][r] *= scale;
      tmax[r] = fmaxf(c[0][r], c[1][r]);
    }
    #pragma unroll
    for (int off = 1; off < 16; off <<= 1)
      #pragma unroll
      for (int r = 0; r < 8; ++r)
        tmax[r] = fmaxf(tmax[r], __shfl_xor(tmax[r], off, 32));
    #pragma unroll
    for (int r = 0; r < 8; ++r) {
      const float mn = fmaxf(mrow[r], tmax[r]);
      alpha[r] = __expf(mrow[r] - mn);
      mrow[r] = mn;
      const float p0 = __expf(c[0][r] - mn);
      const float p1 = __expf(c[1][r] - mn);
      c[0][r] = p0; c[1][r] = p1;
      psum[r] = p0 + p1;
    }
    #pragma unroll
    for (int off = 1; off < 16; off <<= 1)
      #pragma unroll
      for (int r = 0; r < 8; ++r)
        psum[r] += __shfl_xor(psum[r], off, 32);
    #pragma unroll
    for (int r = 0; r < 8; ++r) {
      lrow[r] = lrow[r] * alpha[r] + psum[r];
      o0[r] *= alpha[r]; o1[r] *= alpha[r]; o2[r] *= alpha[r]; o3[r] *= alpha[r];
    }

    // ---- P (C-layout) -> per-wave LDS -> reload as A-fragment ----
    #pragma unroll
    for (int nt = 0; nt < 2; ++nt)
      #pragma unroll
      for (int r = 0; r < 8; ++r)
        Pl[wave][r + 8 * half][nt * 16 + col] = (__bf16)c[nt][r];
    asm volatile("s_wait_dscnt 0" ::: "memory");  // intra-wave DS ordering

    v16bf pa;
    {
      const __bf16* pr = &Pl[wave][col][0];
      #pragma unroll
      for (int g = 0; g < 2; ++g) {
        v8bf r = *(const v8bf*)(pr + g * 16 + h8);
        #pragma unroll
        for (int e = 0; e < 8; ++e) pa[g * 8 + e] = r[e];
      }
    }

    // ---- O += P @ V from LDS V^T tile ----
    #pragma unroll
    for (int t = 0; t < 4; ++t) {
      const __bf16* vr = &Vs[buf][t * 16 + col][h8];
      v16bf bv;
      v8bf r0 = *(const v8bf*)(vr);
      v8bf r1 = *(const v8bf*)(vr + 16);
      #pragma unroll
      for (int e = 0; e < 8; ++e) { bv[e] = r0[e]; bv[8 + e] = r1[e]; }
      v8f& oo = (t == 0) ? o0 : (t == 1) ? o1 : (t == 2) ? o2 : o3;
      oo = wmma_bf16(pa, bv, oo);
    }

    __syncthreads();               // everyone done with buf before re-staging
  }

  #pragma unroll
  for (int r = 0; r < 8; ++r) {
    const float inv = 1.0f / lrow[r];
    o0[r] *= inv; o1[r] *= inv; o2[r] *= inv; o3[r] *= inv;
  }

  // merged-heads bf16 output (B, S, D) for the final dense GEMM
  __bf16* ob = AO + ((size_t)bb * SEQ + qt * 16) * D_MODEL + hh * DEPTH;
  #pragma unroll
  for (int r = 0; r < 8; ++r) {
    const size_t row = (size_t)(r + 8 * half) * D_MODEL;
    ob[row + 0  + col] = (__bf16)o0[r];
    ob[row + 16 + col] = (__bf16)o1[r];
    ob[row + 32 + col] = (__bf16)o2[r];
    ob[row + 48 + col] = (__bf16)o3[r];
  }
}

// ---------------------------------------------------------------------------
extern "C" void kernel_launch(void* const* d_in, const int* in_sizes, int n_in,
                              void* d_out, int out_size, void* d_ws, size_t ws_size,
                              hipStream_t stream) {
  const float* v    = (const float*)d_in[0];
  const float* k    = (const float*)d_in[1];
  const float* q    = (const float*)d_in[2];
  const float* wq_w = (const float*)d_in[3];
  const float* wq_b = (const float*)d_in[4];
  const float* wk_w = (const float*)d_in[5];
  const float* wk_b = (const float*)d_in[6];
  const float* wv_w = (const float*)d_in[7];
  const float* wv_b = (const float*)d_in[8];
  const float* dw   = (const float*)d_in[9];
  const float* db   = (const float*)d_in[10];
  const float* pe   = (const float*)d_in[11];

  char* ws = (char*)d_ws;
  const size_t SZ = (size_t)16 * 1024 * 1024;   // 16 MB per bf16 tensor
  __bf16* Qh = (__bf16*)(ws + 0 * SZ);          // (B,H,S,64)
  __bf16* Kh = (__bf16*)(ws + 1 * SZ);          // (B,H,S,64)
  __bf16* Vt = (__bf16*)(ws + 2 * SZ);          // (B,H,64,S)
  __bf16* AO = (__bf16*)(ws + 3 * SZ);          // (B,S,D)

  dim3 blk(256);
  dim3 grd(D_MODEL / 512, (4 * SEQ) / 16);      // (2, 512)

  proj_gemm<true, 0><<<grd, blk, 0, stream>>>(q, pe, wq_w, wq_b, Qh);
  proj_gemm<true, 0><<<grd, blk, 0, stream>>>(k, pe, wk_w, wk_b, Kh);
  proj_gemm<true, 1><<<grd, blk, 0, stream>>>(v, pe, wv_w, wv_b, Vt);

  flash_attn<<<dim3(1024), blk, 0, stream>>>(Qh, Kh, Vt, AO);

  proj_gemm<false, 2><<<grd, blk, 0, stream>>>(AO, nullptr, dw, db, d_out);
}
